// CPGaussian_19275813224808
// MI455X (gfx1250) — matrix operand order repaired and checked by reading, not compiled, not run on hardware
//
#include <hip/hip_runtime.h>
#include <hip/hip_bf16.h>

typedef __attribute__((ext_vector_type(16))) __bf16 v16bf;
typedef __attribute__((ext_vector_type(8)))  __bf16 v8bf;
typedef __attribute__((ext_vector_type(8)))  float  v8f;
typedef __attribute__((ext_vector_type(4)))  float  v4f;
typedef __attribute__((ext_vector_type(4)))  unsigned int u32x4;
typedef __attribute__((ext_vector_type(4)))  int    i32x4;
typedef __attribute__((ext_vector_type(8)))  int    i32x8;

#define NFEAT 128            // M: gaussian dimensionality
#define NCOMP 256            // K: mixture components
#define KDIM  256            // GEMM reduction dim = 2*NFEAT ([x^2, x])
#define LOG_2PI 1.8378770664093453f
#define ROWS_PER_BLOCK 128   // 8 waves x 16 rows (single strip: no spills)

#ifndef __has_builtin
#define __has_builtin(x) 0
#endif
#if __has_builtin(__builtin_amdgcn_tensor_load_to_lds) && \
    __has_builtin(__builtin_amdgcn_s_wait_tensorcnt)
#define HAVE_TDM 1
#else
#define HAVE_TDM 0
#endif

// ---------------------------------------------------------------------------
// Prep kernel 1: one block per component k, one thread per feature m.
// Builds the fused coefficient matrix B, stored TRANSPOSED (Bt[f][k]) so the
// WMMA B-fragment (lane%16 = N/comp index, vector elem = K/feature index) is a
// per-element 32B LDS read contiguous across 16 lanes. Stored as bf16 hi + bf16 residual (lo) for the
// split-precision GEMM. Also accumulates the per-component bias (raw w_logits added
// here; prep kernel 2 subtracts their logsumexp).
// ---------------------------------------------------------------------------
__global__ void __launch_bounds__(128)
prep_B_kernel(const float* __restrict__ locs, const float* __restrict__ scales,
              const float* __restrict__ w_logits,
              __bf16* __restrict__ Bth, __bf16* __restrict__ Btl,
              float* __restrict__ bias0)
{
    __shared__ float red[128];
    const int k = blockIdx.x;     // component
    const int m = threadIdx.x;    // feature
    const float sc = scales[(size_t)k * NFEAT + m];
    const float mu = locs  [(size_t)k * NFEAT + m];
    const float iv = 1.0f / (sc * sc);

    const float c0 = -0.5f * iv;  // coefficient of x^2  (GEMM feature m)
    const float c1 = mu * iv;     // coefficient of x    (GEMM feature 128+m)

    __bf16 h0 = (__bf16)c0;
    Bth[(size_t)m * NCOMP + k] = h0;
    Btl[(size_t)m * NCOMP + k] = (__bf16)(c0 - (float)h0);
    __bf16 h1 = (__bf16)c1;
    Bth[(size_t)(NFEAT + m) * NCOMP + k] = h1;
    Btl[(size_t)(NFEAT + m) * NCOMP + k] = (__bf16)(c1 - (float)h1);

    // bias contribution: -0.5*mu^2/var - log(sigma)
    red[m] = -0.5f * mu * mu * iv - logf(sc);
    __syncthreads();
#pragma unroll
    for (int off = 64; off > 0; off >>= 1) {
        if (m < off) red[m] += red[m + off];
        __syncthreads();
    }
    if (m == 0)
        bias0[k] = red[0] - 0.5f * LOG_2PI * (float)NFEAT + w_logits[k];
}

// ---------------------------------------------------------------------------
// Prep kernel 2: single block; log-softmax normalization of w_logits folded
// into the bias.
// ---------------------------------------------------------------------------
__global__ void __launch_bounds__(256)
prep_bias_kernel(const float* __restrict__ w_logits,
                 const float* __restrict__ bias0,
                 float* __restrict__ bias)
{
    __shared__ float red[256];
    const int t = threadIdx.x;
    const float v = w_logits[t];
    red[t] = v;
    __syncthreads();
#pragma unroll
    for (int off = 128; off > 0; off >>= 1) {
        if (t < off) red[t] = fmaxf(red[t], red[t + off]);
        __syncthreads();
    }
    const float mx = red[0];
    __syncthreads();
    red[t] = expf(v - mx);
    __syncthreads();
#pragma unroll
    for (int off = 128; off > 0; off >>= 1) {
        if (t < off) red[t] += red[t + off];
        __syncthreads();
    }
    const float lse = mx + logf(red[0]);
    bias[t] = bias0[t] - lse;
}

#if HAVE_TDM
// ---------------------------------------------------------------------------
// Issue one TDM descriptor copying `bytes` (multiple of 8) from global memory
// into LDS at byte offset `lds_off`. 1-D tensor of 8-byte elements, per the
// CDNA5 D# layout (group0: count/lds/global/type, group1: data_size + dims).
// Tracked on TENSORcnt; EXEC-independent, so issue from a single wave.
// ---------------------------------------------------------------------------
__device__ __forceinline__ void tdm_copy_to_lds(const void* gsrc,
                                                unsigned lds_off,
                                                unsigned bytes)
{
    const unsigned n8 = bytes >> 3;                    // 8-byte elements
    const unsigned long long ga = (unsigned long long)(size_t)gsrc;

    u32x4 g0;
    g0.x = 1u;                                         // count=1 (valid, user)
    g0.y = lds_off;                                    // lds_addr [63:32]
    g0.z = (unsigned)(ga & 0xffffffffu);               // global_addr[31:0]
    g0.w = (unsigned)((ga >> 32) & 0x01ffffffu)        // global_addr[56:32]
         | (2u << 30);                                 // type=2 ("image")

    i32x8 g1;
    g1[0] = (int)(3u << 16);                           // data_size=3 (8B)
    g1[1] = (int)((n8 & 0xffffu) << 16);               // tensor_dim0[15:0]@48
    g1[2] = (int)(((n8 >> 16) & 0xffffu)               // tensor_dim0[31:16]
         |        (1u << 16));                         // tensor_dim1=1 @80
    g1[3] = (int)((n8 & 0xffffu) << 16);               // tile_dim0 @112
    g1[4] = 1;                                         // tile_dim1=1 @128
    g1[5] = (int)n8;                                   // tensor_dim0_stride lo
    g1[6] = 0;                                         // stride0 hi / stride1 lo
    g1[7] = 0;                                         // stride1 hi

    i32x4 z4 = {0, 0, 0, 0};
#if (__clang_major__ >= 23)
    i32x8 z8 = {0, 0, 0, 0, 0, 0, 0, 0};
    __builtin_amdgcn_tensor_load_to_lds(g0, g1, z4, z4, z8, 0);
#else
    __builtin_amdgcn_tensor_load_to_lds(g0, g1, z4, z4, 0);
#endif
}
#endif // HAVE_TDM

// ---------------------------------------------------------------------------
// Main kernel: 256 threads = 8 wave32's; each wave owns one 16-row strip
// (block covers 128 rows). All of B (hi+lo, 256 KB) lives in LDS (320 KB/WGP),
// staged by the Tensor Data Mover. Split-precision bf16 GEMM:
// acc += Ah*Bh + Ah*Bl + Al*Bh (3 WMMAs/tile), fused bias add + row-wise
// logsumexp epilogue done entirely in registers.
// ---------------------------------------------------------------------------
__global__ void __launch_bounds__(256)
mixture_wmma_kernel(const float* __restrict__ x,
                    const __bf16* __restrict__ gBh,
                    const __bf16* __restrict__ gBl,
                    const float* __restrict__ bias,
                    float* __restrict__ out)
{
    extern __shared__ __bf16 smem[];                 // [2][KDIM][NCOMP] + bias
    __bf16* sBh   = smem;
    __bf16* sBl   = smem + (size_t)KDIM * NCOMP;
    float*  sBias = (float*)(smem + (size_t)2 * KDIM * NCOMP);

    const int tid  = threadIdx.x;
    const int lane = tid & 31;
    const int wave = tid >> 5;
    const unsigned bmat_bytes = (unsigned)(KDIM * NCOMP * 2);   // 131072

#if HAVE_TDM
    if (wave == 0) {
        tdm_copy_to_lds(gBh, 0u, bmat_bytes);
        tdm_copy_to_lds(gBl, bmat_bytes, bmat_bytes);
        __builtin_amdgcn_s_wait_tensorcnt(0);
    }
#else
    {
        const int total = (KDIM * NCOMP) / 8;        // 8192 v8bf chunks
        const v8bf* sh = (const v8bf*)gBh;
        v8bf*       dh = (v8bf*)sBh;
        for (int i = tid; i < total; i += 256) dh[i] = sh[i];
        const v8bf* sl = (const v8bf*)gBl;
        v8bf*       dl = (v8bf*)sBl;
        for (int i = tid; i < total; i += 256) dl[i] = sl[i];
    }
#endif
    for (int i = tid; i < NCOMP; i += 256) sBias[i] = bias[i];
    __syncthreads();

    const int halfHi  = (lane >> 4) & 1;     // 0: lanes 0-15, 1: lanes 16-31
    const int rowBase = blockIdx.x * ROWS_PER_BLOCK + wave * 16;
    const float* xrow = x + (size_t)(rowBase + (lane & 15)) * NFEAT;
    const int r0 = halfHi ? 8  : 0;          // A-fragment K-offset runs
    const int r1 = halfHi ? 24 : 16;

    v8f acc[16];
#pragma unroll
    for (int ct = 0; ct < 16; ++ct)
#pragma unroll
        for (int j = 0; j < 8; ++j) acc[ct][j] = 0.0f;

#pragma unroll
    for (int kt = 0; kt < 8; ++kt) {
        // GEMM features kt*32..+31; kt<4 -> x^2 terms, kt>=4 -> x terms
        const int  xbase  = (kt & 3) * 32;
        const bool square = (kt < 4);
        v4f xa = *(const v4f*)(xrow + xbase + r0);
        v4f xb = *(const v4f*)(xrow + xbase + r0 + 4);
        v4f xc = *(const v4f*)(xrow + xbase + r1);
        v4f xd = *(const v4f*)(xrow + xbase + r1 + 4);
        float vals[16];
#pragma unroll
        for (int i = 0; i < 4; ++i) {
            vals[i]      = xa[i];
            vals[4 + i]  = xb[i];
            vals[8 + i]  = xc[i];
            vals[12 + i] = xd[i];
        }
        v16bf a_hi, a_lo;
#pragma unroll
        for (int i = 0; i < 16; ++i) {
            float v = vals[i];
            if (square) v = v * v;
            __bf16 h = (__bf16)v;
            a_hi[i] = h;
            a_lo[i] = (__bf16)(v - (float)h);
        }
        // B fragment: lane%16 indexes N (component), elems index K (feature, same runs as A)
        const __bf16* bhrow = sBh + (size_t)(kt * 32) * NCOMP + (lane & 15);
        const __bf16* blrow = sBl + (size_t)(kt * 32) * NCOMP + (lane & 15);
#pragma unroll
        for (int ct = 0; ct < 16; ++ct) {
            v16bf bh, bl;
#pragma unroll
            for (int s = 0; s < 16; ++s) {
                const int kk = (s < 8) ? (r0 + s) : (r1 + s - 8);
                bh[s] = bhrow[(size_t)kk * NCOMP + ct * 16];
                bl[s] = blrow[(size_t)kk * NCOMP + ct * 16];
            }
            acc[ct] = __builtin_amdgcn_wmma_f32_16x16x32_bf16(
                false, a_hi, false, bh, (short)0, acc[ct], false, false);
            acc[ct] = __builtin_amdgcn_wmma_f32_16x16x32_bf16(
                false, a_lo, false, bh, (short)0, acc[ct], false, false);
            acc[ct] = __builtin_amdgcn_wmma_f32_16x16x32_bf16(
                false, a_hi, false, bl, (short)0, acc[ct], false, false);
        }
    }

    // Per-component bias (column = ct*16 + lane%16, same for all 8 rows).
#pragma unroll
    for (int ct = 0; ct < 16; ++ct) {
        const float b = sBias[ct * 16 + (lane & 15)];
#pragma unroll
        for (int j = 0; j < 8; ++j) acc[ct][j] += b;
    }

    // Row-wise logsumexp. C layout: VGPR j holds row (j + 8*halfHi) at column
    // lane%16; reduce in-register over ct, then across the 16-lane group.
    float mrow[8], srow[8];
#pragma unroll
    for (int j = 0; j < 8; ++j) {
        float mx = acc[0][j];
#pragma unroll
        for (int ct = 1; ct < 16; ++ct) mx = fmaxf(mx, acc[ct][j]);
#pragma unroll
        for (int d = 1; d < 16; d <<= 1)
            mx = fmaxf(mx, __shfl_xor(mx, d, 32));
        float sum = 0.0f;
#pragma unroll
        for (int ct = 0; ct < 16; ++ct) sum += __expf(acc[ct][j] - mx);
#pragma unroll
        for (int d = 1; d < 16; d <<= 1)
            sum += __shfl_xor(sum, d, 32);
        mrow[j] = mx;
        srow[j] = sum;
    }
    if ((lane & 15) == 0) {
        const int rb = rowBase + halfHi * 8;
#pragma unroll
        for (int j = 0; j < 8; ++j) out[rb + j] = mrow[j] + logf(srow[j]);
    }
}

// ---------------------------------------------------------------------------
extern "C" void kernel_launch(void* const* d_in, const int* in_sizes, int n_in,
                              void* d_out, int out_size, void* d_ws, size_t ws_size,
                              hipStream_t stream)
{
    const float* x        = (const float*)d_in[0];   // [N, 128]
    const float* w_logits = (const float*)d_in[1];   // [256]
    const float* locs     = (const float*)d_in[2];   // [256, 128]
    const float* scales   = (const float*)d_in[3];   // [256, 128]
    float* out = (float*)d_out;                      // [N]

    const int N = in_sizes[0] / NFEAT;               // 131072

    char* ws = (char*)d_ws;
    const size_t bmat_bytes = (size_t)KDIM * NCOMP * sizeof(__bf16);  // 128 KB
    __bf16* Bth   = (__bf16*)(ws);
    __bf16* Btl   = (__bf16*)(ws + bmat_bytes);
    float*  bias0 = (float*) (ws + 2 * bmat_bytes);
    float*  bias  = (float*) (ws + 2 * bmat_bytes + 1024);

    prep_B_kernel<<<NCOMP, NFEAT, 0, stream>>>(locs, scales, w_logits,
                                               Bth, Btl, bias0);
    prep_bias_kernel<<<1, NCOMP, 0, stream>>>(w_logits, bias0, bias);

    const size_t shmem = 2 * bmat_bytes + NCOMP * sizeof(float); // 257 KB
    const int blocks = N / ROWS_PER_BLOCK;                       // 1024
    mixture_wmma_kernel<<<blocks, 256, shmem, stream>>>(x, Bth, Btl, bias, out);
}
